// ColumnSparseMLP_74629351735482
// MI455X (gfx1250) — matrix operand order repaired and checked by reading, not compiled, not run on hardware
//
#include <hip/hip_runtime.h>
#include <hip/hip_bf16.h>

#define N_TOKENS 4096
#define D_IN     4096
#define D_HID    16384
#define D_OUT    4096
#define K_NZ     8192

typedef __bf16 bf16_t;
typedef bf16_t v16bf __attribute__((ext_vector_type(16)));
typedef bf16_t v8bf  __attribute__((ext_vector_type(8)));
typedef bf16_t v4bf  __attribute__((ext_vector_type(4)));
typedef float  v8f   __attribute__((ext_vector_type(8)));
typedef int    v4i_t __attribute__((ext_vector_type(4)));

union FragU { v16bf v; v8bf h[2]; };

#if __has_builtin(__builtin_amdgcn_global_load_async_to_lds_b128)
#define USE_ASYNC_LDS 1
// builtin signature (from clang diagnostic): (v4i AS1*, v4i AS3*, imm offset, imm cpol)
typedef __attribute__((address_space(1))) v4i_t gv4i;
typedef __attribute__((address_space(3))) v4i_t lv4i;
#else
#define USE_ASYNC_LDS 0
#endif

__device__ __forceinline__ void wait_async_zero() {
#if __has_builtin(__builtin_amdgcn_s_wait_asynccnt)
  __builtin_amdgcn_s_wait_asynccnt(0);
#else
  asm volatile("s_wait_asynccnt 0x0" ::: "memory");
#endif
}

// ---------------- preprocessing: f32 -> bf16 convert ----------------
__global__ __launch_bounds__(256) void cvt_bf16_kernel(
    const float* __restrict__ src, bf16_t* __restrict__ dst, int n) {
  int i = (blockIdx.x * 256 + threadIdx.x) * 4;
  if (i + 3 < n) {
    float4 f = *(const float4*)(src + i);
    v4bf o;
    o.x = (bf16_t)f.x; o.y = (bf16_t)f.y; o.z = (bf16_t)f.z; o.w = (bf16_t)f.w;
    *(v4bf*)(dst + i) = o;
  }
}

// ---------------- preprocessing: gather fc1 rows -> bf16 ----------------
// w1b[k, i] = bf16(fc1[nz[k], i]),  shape K_NZ x D_IN, row-major
__global__ __launch_bounds__(256) void gather_rows_kernel(
    const float* __restrict__ w, const int* __restrict__ idx, bf16_t* __restrict__ dst) {
  int k = blockIdx.x;
  int row = idx[k];
  const float* src = w + (size_t)row * D_IN;
  bf16_t* d = dst + (size_t)k * D_IN;
  for (int i = threadIdx.x * 4; i < D_IN; i += 256 * 4) {
    float4 f = *(const float4*)(src + i);
    v4bf o;
    o.x = (bf16_t)f.x; o.y = (bf16_t)f.y; o.z = (bf16_t)f.z; o.w = (bf16_t)f.w;
    *(v4bf*)(d + i) = o;
  }
}

// ---------------- preprocessing: gather fc2 columns -> bf16 ----------------
// w2b[o, k] = bf16(fc2[o, nz[k]]),  shape D_OUT x K_NZ, row-major (k contiguous)
__global__ __launch_bounds__(256) void gather_cols_kernel(
    const float* __restrict__ w2, const int* __restrict__ idx, bf16_t* __restrict__ dst) {
  int k   = blockIdx.x * 256 + threadIdx.x;   // 0..K_NZ-1
  int col = idx[k];
  int o0  = blockIdx.y * 8;
#pragma unroll
  for (int j = 0; j < 8; ++j) {
    int o = o0 + j;
    dst[(size_t)o * K_NZ + k] = (bf16_t)w2[(size_t)o * D_HID + col];
  }
}

// ---------------- bf16 NT-GEMM: C[m,n] = sum_k A[m,k]*B[n,k] ----------------
// A: M x Kd row-major bf16, B: N x Kd row-major bf16.
// WG = 256 threads (8 waves as 4x2), tile 256x128, wave tile 64x64 (4x4 WMMA),
// K-step 32, double-buffered LDS fed by GLOBAL_LOAD_ASYNC_TO_LDS_B128.
#define LDSS 40   // padded LDS row stride (elements): 80B spreads b128 bank groups

template <bool OUT_BF16>
__global__ __launch_bounds__(256) void gemm_nt_bf16(
    const bf16_t* __restrict__ A, const bf16_t* __restrict__ B,
    void* __restrict__ C, int M, int N, int Kd) {
  __shared__ bf16_t sA[2][256 * LDSS];   // 40 KB
  __shared__ bf16_t sB[2][128 * LDSS];   // 20 KB

  const int tid   = threadIdx.x;
  const int lane  = tid & 31;
  const int wid   = tid >> 5;
  const int wm    = wid >> 1;      // 0..3  (64 rows each)
  const int wn    = wid & 1;       // 0..1  (64 cols each)
  const int l16   = lane & 15;
  const int lhalf = lane >> 4;

  const size_t blockM = (size_t)blockIdx.y * 256;
  const size_t blockN = (size_t)blockIdx.x * 128;

  const bf16_t* gA = A + blockM * (size_t)Kd;
  const bf16_t* gB = B + blockN * (size_t)Kd;

  // 16B chunk decomposition of the K-step tiles:
  //   A: 256 rows x 64B = 1024 chunks -> 4 per thread
  //   B: 128 rows x 64B =  512 chunks -> 2 per thread
#if USE_ASYNC_LDS
  auto stageTile = [&](int t, int buf) {
    size_t k0 = (size_t)t * 32;
#pragma unroll
    for (int j = 0; j < 4; ++j) {
      int c = tid + 256 * j;
      int r = c >> 2, col = (c & 3) * 8;
      __builtin_amdgcn_global_load_async_to_lds_b128(
          (gv4i*)(uintptr_t)(gA + (size_t)r * Kd + k0 + col),
          (lv4i*)(unsigned)(uintptr_t)&sA[buf][r * LDSS + col], 0, 0);
    }
#pragma unroll
    for (int j = 0; j < 2; ++j) {
      int c = tid + 256 * j;
      int r = c >> 2, col = (c & 3) * 8;
      __builtin_amdgcn_global_load_async_to_lds_b128(
          (gv4i*)(uintptr_t)(gB + (size_t)r * Kd + k0 + col),
          (lv4i*)(unsigned)(uintptr_t)&sB[buf][r * LDSS + col], 0, 0);
    }
  };
#else
  uint4 pa[4], pb[2];
  auto loadG = [&](int t) {
    size_t k0 = (size_t)t * 32;
#pragma unroll
    for (int j = 0; j < 4; ++j) {
      int c = tid + 256 * j;
      int r = c >> 2, col = (c & 3) * 8;
      pa[j] = *(const uint4*)(gA + (size_t)r * Kd + k0 + col);
    }
#pragma unroll
    for (int j = 0; j < 2; ++j) {
      int c = tid + 256 * j;
      int r = c >> 2, col = (c & 3) * 8;
      pb[j] = *(const uint4*)(gB + (size_t)r * Kd + k0 + col);
    }
  };
  auto storeL = [&](int buf) {
#pragma unroll
    for (int j = 0; j < 4; ++j) {
      int c = tid + 256 * j;
      int r = c >> 2, col = (c & 3) * 8;
      *(uint4*)&sA[buf][r * LDSS + col] = pa[j];
    }
#pragma unroll
    for (int j = 0; j < 2; ++j) {
      int c = tid + 256 * j;
      int r = c >> 2, col = (c & 3) * 8;
      *(uint4*)&sB[buf][r * LDSS + col] = pb[j];
    }
  };
#endif

  v8f acc[4][4] = {};

  // prologue: stage K-tile 0 into buffer 0
#if USE_ASYNC_LDS
  stageTile(0, 0);
  wait_async_zero();
#else
  loadG(0);
  storeL(0);
#endif
  __syncthreads();

  const int kTiles = Kd / 32;
  for (int t = 0; t < kTiles; ++t) {
    const int cur = t & 1;
#if USE_ASYNC_LDS
    if (t + 1 < kTiles) stageTile(t + 1, 1 - cur);  // async, overlaps compute
#else
    if (t + 1 < kTiles) loadG(t + 1);
#endif

    // B fragments (32x16 bf16 layout: lanes = N col, half-lanes split K 0-15/16-31)
    FragU bfrag[4];
#pragma unroll
    for (int nj = 0; nj < 4; ++nj) {
      int row = wn * 64 + nj * 16 + l16;
      int off = row * LDSS + lhalf * 16;
      bfrag[nj].h[0] = *(const v8bf*)&sB[cur][off];
      bfrag[nj].h[1] = *(const v8bf*)&sB[cur][off + 8];
    }
    // stream A fragments (16x32 bf16 layout), 4 WMMAs each
#pragma unroll
    for (int mi = 0; mi < 4; ++mi) {
      FragU af;
      int row = wm * 64 + mi * 16 + l16;
      int off = row * LDSS + lhalf * 16;
      af.h[0] = *(const v8bf*)&sA[cur][off];        // K = base..base+7
      af.h[1] = *(const v8bf*)&sA[cur][off + 8];    // K = base+8..base+15
#pragma unroll
      for (int nj = 0; nj < 4; ++nj)
        acc[mi][nj] = __builtin_amdgcn_wmma_f32_16x16x32_bf16(
            false, af.v, false, bfrag[nj].v, (short)0, acc[mi][nj],
            false, false);
    }

#if USE_ASYNC_LDS
    if (t + 1 < kTiles) wait_async_zero();
#else
    if (t + 1 < kTiles) storeL(1 - cur);
#endif
    __syncthreads();
  }

  // epilogue: C/D layout = 8 VGPRs, N = lane%16, M = reg + 8*(lane/16)
  const size_t cm = blockM + wm * 64;
  const size_t cn = blockN + wn * 64;
#pragma unroll
  for (int mi = 0; mi < 4; ++mi) {
#pragma unroll
    for (int nj = 0; nj < 4; ++nj) {
      size_t col = cn + nj * 16 + l16;
#pragma unroll
      for (int r = 0; r < 8; ++r) {
        size_t row = cm + mi * 16 + r + lhalf * 8;
        if (OUT_BF16)
          ((bf16_t*)C)[row * (size_t)N + col] = (bf16_t)acc[mi][nj][r];
        else
          ((float*)C)[row * (size_t)N + col] = acc[mi][nj][r];
      }
    }
  }
}

// ---------------- launcher ----------------
extern "C" void kernel_launch(void* const* d_in, const int* in_sizes, int n_in,
                              void* d_out, int out_size, void* d_ws, size_t ws_size,
                              hipStream_t stream) {
  const float* x   = (const float*)d_in[0];
  const int*   nz  = (const int*)d_in[1];
  const float* fc1 = (const float*)d_in[2];
  const float* fc2 = (const float*)d_in[3];
  float* out = (float*)d_out;

  char* ws = (char*)d_ws;
  bf16_t* xb  = (bf16_t*)(ws);                   //  32 MB: x in bf16
  bf16_t* w1b = (bf16_t*)(ws + (32ull  << 20));  //  64 MB: gathered fc1 rows
  bf16_t* hb  = (bf16_t*)(ws + (96ull  << 20));  //  64 MB: hidden activations (bf16)
  bf16_t* w2b = (bf16_t*)(ws + (160ull << 20));  //  64 MB: gathered fc2 columns

  {
    int n = N_TOKENS * D_IN;
    cvt_bf16_kernel<<<n / (256 * 4), 256, 0, stream>>>(x, xb, n);
  }
  gather_rows_kernel<<<K_NZ, 256, 0, stream>>>(fc1, nz, w1b);
  gather_cols_kernel<<<dim3(K_NZ / 256, D_OUT / 8), 256, 0, stream>>>(fc2, nz, w2b);

  // GEMM1: h = x @ w1^T   (M=4096, N=8192, K=4096), bf16 output
  gemm_nt_bf16<true><<<dim3(K_NZ / 128, N_TOKENS / 256), 256, 0, stream>>>(
      xb, w1b, (void*)hb, N_TOKENS, K_NZ, D_IN);

  // GEMM2: out = h @ w2^T (M=4096, N=4096, K=8192), f32 output
  gemm_nt_bf16<false><<<dim3(D_OUT / 128, N_TOKENS / 256), 256, 0, stream>>>(
      hb, w2b, (void*)out, N_TOKENS, D_OUT, K_NZ);
}
